// SympleAgent_43215960932510
// MI455X (gfx1250) — matrix operand order repaired and checked by reading, not compiled, not run on hardware
//
#include <hip/hip_runtime.h>
#include <hip/hip_bf16.h>

#define HDIM   768
#define NNODES 16383
#define DMAX   13
#define GAS    __attribute__((address_space(1)))

typedef __attribute__((ext_vector_type(16))) __bf16 bf16x16;
typedef __attribute__((ext_vector_type(8)))  float  f32x8;
typedef __attribute__((ext_vector_type(4)))  unsigned int u32x4;

union ABFrag { u32x4 u[2]; bf16x16 v; };

// explicit global-address-space accessors: lower to global_load/global_store
// (not flat_*), so they are tracked by LOADcnt/STOREcnt only, never DScnt.
template <typename T>
__device__ __forceinline__ T ldg(const T* p) {
  return *(const GAS T*)(unsigned long long)p;
}
template <typename T>
__device__ __forceinline__ void stg(T* p, T v) {
  *(GAS T*)(unsigned long long)p = v;
}

__device__ __forceinline__ f32x8 wmma_bf16(bf16x16 a, bf16x16 b, f32x8 c) {
  return __builtin_amdgcn_wmma_f32_16x16x32_bf16(false, a, false, b, (short)0, c, false, false);
}

// A fragment from LDS: lane l=lane%16 is row; e<8 -> k=k0+hi*8+e, e>=8 -> k=k0+16+hi*8+(e-8)
__device__ __forceinline__ bf16x16 load_afrag(const __bf16* row, int k0, int hi) {
  ABFrag f;
  f.u[0] = *(const u32x4*)(row + k0 + hi * 8);
  f.u[1] = *(const u32x4*)(row + k0 + 16 + hi * 8);
  return f.v;
}
// B fragment: 16 contiguous bf16 per lane from the pre-packed global buffer
__device__ __forceinline__ bf16x16 load_bfrag(const __bf16* p) {
  ABFrag f;
  const GAS u32x4* gp = (const GAS u32x4*)(unsigned long long)p;
  f.u[0] = gp[0];
  f.u[1] = gp[1];
  return f.v;
}

__device__ __forceinline__ float sigm(float x)      { return 1.f / (1.f + __expf(-x)); }
__device__ __forceinline__ float tanh_fast(float x) { return 2.f / (1.f + __expf(-2.f * x)) - 1.f; }

// ---------------- weight packing ----------------
// packed index p = ((kstep*ntiles + ntile)*32 + lane)*16 + e
// value = src[k*O + o], k = kstep*32 + (e/8)*16 + (lane/16)*8 + (e%8), o = ntile*16 + (lane%16)
__global__ void __launch_bounds__(256) pack_w(const float* __restrict__ src,
                                              __bf16* __restrict__ dst, int K, int O) {
  long p = (long)blockIdx.x * 256 + threadIdx.x;
  if (p >= (long)K * O) return;
  int e = (int)(p & 15);
  int lane = (int)((p >> 4) & 31);
  int rem = (int)(p >> 9);
  int ntiles = O >> 4;
  int ntile = rem % ntiles;
  int kstep = rem / ntiles;
  int k = kstep * 32 + (e >> 3) * 16 + (lane >> 4) * 8 + (e & 7);
  int o = ntile * 16 + (lane & 15);
  stg(dst + p, (__bf16)ldg(src + (size_t)k * O + o));
}

// heads: pack [hl_W(768x2) | int_W(768x2) | act_W(768x64) | zero-pad] -> 768x80
__global__ void __launch_bounds__(256) pack_heads(const float* __restrict__ hlW,
                                                  const float* __restrict__ inW,
                                                  const float* __restrict__ acW,
                                                  __bf16* __restrict__ dst) {
  int p = blockIdx.x * 256 + threadIdx.x;
  if (p >= 768 * 80) return;
  int e = p & 15;
  int lane = (p >> 4) & 31;
  int rem = p >> 9;
  int ntile = rem % 5;
  int kstep = rem / 5;
  int k = kstep * 32 + (e >> 3) * 16 + (lane >> 4) * 8 + (e & 7);
  int o = ntile * 16 + (lane & 15);
  float v = 0.f;
  if (o < 2)       v = ldg(hlW + k * 2 + o);
  else if (o < 4)  v = ldg(inW + k * 2 + (o - 2));
  else if (o < 68) v = ldg(acW + k * 64 + (o - 4));
  stg(dst + p, (__bf16)v);
}

// ---------------- pass 1: binary Tree-LSTM, one launch per level ----------------
__global__ void __launch_bounds__(256) cell_pass1(
    const int* __restrict__ ids,
    const float* __restrict__ Wb_iou, const float* __restrict__ Wb_f,
    const float* __restrict__ bb_iou, const float* __restrict__ bb_f,
    const __bf16* __restrict__ PWiou, const __bf16* __restrict__ PWf0,
    const __bf16* __restrict__ PWf1,
    __bf16* __restrict__ h_bf, float* __restrict__ c_f32, int d) {
  extern __shared__ __align__(16) char smem[];
  __bf16* ldsA = (__bf16*)smem;
  const int LDA = 1536 + 8;
  const int levelN = 1 << d;
  const int off = levelN - 1;
  const int tile = blockIdx.x;
  const int tid = threadIdx.x;
  const int wave = tid >> 5, lane = tid & 31;
  const int l = lane & 15, hi = lane >> 4;
  const bool leaf = (d == DMAX);
  const int choff = leaf ? 0 : ((1 << (d + 1)) - 1);

  if (!leaf) {  // stage A = [h_left | h_right] (children are consecutive rows)
    for (int idx = tid; idx < 16 * 192; idx += 256) {
      int r = idx / 192, k = (idx - r * 192) * 8;
      int node = tile * 16 + r;
      u32x4 val = {0u, 0u, 0u, 0u};
      if (node < levelN) val = ldg((const u32x4*)(h_bf + (size_t)(choff + 2 * node) * HDIM + k));
      *(u32x4*)(ldsA + r * LDA + k) = val;
    }
  }
  __syncthreads();

  for (int jt = wave; jt < 48; jt += 8) {
    f32x8 acc[5];
#pragma unroll
    for (int s = 0; s < 5; ++s) { f32x8 z = {0.f,0.f,0.f,0.f,0.f,0.f,0.f,0.f}; acc[s] = z; }
    if (!leaf) {
      const __bf16* arow = ldsA + l * LDA;
      const __bf16* bp[5];
      size_t bstr[5];
      bp[0] = PWiou + (size_t)(jt)       * 512 + lane * 16; bstr[0] = (size_t)144 * 512;
      bp[1] = PWiou + (size_t)(48 + jt)  * 512 + lane * 16; bstr[1] = (size_t)144 * 512;
      bp[2] = PWiou + (size_t)(96 + jt)  * 512 + lane * 16; bstr[2] = (size_t)144 * 512;
      bp[3] = PWf0  + (size_t)jt         * 512 + lane * 16; bstr[3] = (size_t)48 * 512;
      bp[4] = PWf1  + (size_t)jt         * 512 + lane * 16; bstr[4] = (size_t)48 * 512;
      for (int ks = 0; ks < 48; ++ks) {
        bf16x16 a = load_afrag(arow, ks * 32, hi);
#pragma unroll
        for (int s = 0; s < 5; ++s) {
          acc[s] = wmma_bf16(a, load_bfrag(bp[s]), acc[s]);
          bp[s] += bstr[s];
        }
      }
    }
    const int j = jt * 16 + l;
    const float bi = ldg(bb_iou + j), bo = ldg(bb_iou + HDIM + j), bg = ldg(bb_iou + 2 * HDIM + j);
    const float bfv = ldg(bb_f + j);
#pragma unroll
    for (int r = 0; r < 8; ++r) {
      int m = r + hi * 8;
      int node = tile * 16 + m;
      if (node < levelN) {
        int gid = off + node;
        int lid = ldg(ids + gid);
        const float* xr = Wb_iou + (size_t)lid * (3 * HDIM);
        float xf = ldg(Wb_f + (size_t)lid * HDIM + j);
        float ig = sigm(acc[0][r] + ldg(xr + j) + bi);
        float og = sigm(acc[1][r] + ldg(xr + HDIM + j) + bo);
        float gg = tanh_fast(acc[2][r] + ldg(xr + 2 * HDIM + j) + bg);
        float f0 = sigm(acc[3][r] + xf + bfv);
        float f1 = sigm(acc[4][r] + xf + bfv);
        float cc0 = 0.f, cc1 = 0.f;
        if (!leaf) {
          cc0 = ldg(c_f32 + (size_t)(choff + 2 * node) * HDIM + j);
          cc1 = ldg(c_f32 + (size_t)(choff + 2 * node + 1) * HDIM + j);
        }
        float cv = ig * gg + f0 * cc0 + f1 * cc1;
        float hv = og * tanh_fast(cv);
        stg(h_bf + (size_t)gid * HDIM + j, (__bf16)hv);
        stg(c_f32 + (size_t)gid * HDIM + j, cv);
      }
    }
  }
}

// ---------------- pass 2: ternary cell over all levels, single launch ----------------
__global__ void __launch_bounds__(256) cell_pass2(
    const int* __restrict__ ids,
    const float* __restrict__ Wt_iou, const float* __restrict__ Wt_f,
    const float* __restrict__ bt_iou, const float* __restrict__ bt_f,
    const __bf16* __restrict__ PWiou, const __bf16* __restrict__ PWf0,
    const __bf16* __restrict__ PWf1, const __bf16* __restrict__ PWf2,
    const __bf16* __restrict__ h_bf, const float* __restrict__ c_f32,
    __bf16* __restrict__ h3_bf) {
  extern __shared__ __align__(16) char smem[];
  __bf16* ldsA = (__bf16*)smem;
  const int LDA = 2304 + 8;

  int b = blockIdx.x, d = 0, tile = 0;
  for (int dd = 0; dd <= DMAX; ++dd) {
    int t = ((1 << dd) + 15) >> 4;
    if (b < t) { d = dd; tile = b; break; }
    b -= t;
  }
  const int levelN = 1 << d;
  const int off = levelN - 1;
  const int tid = threadIdx.x;
  const int wave = tid >> 5, lane = tid & 31;
  const int l = lane & 15, hi = lane >> 4;
  const bool leaf = (d == DMAX);
  const int choff = leaf ? 0 : ((1 << (d + 1)) - 1);

  if (!leaf) {  // stage A = [self | child0 | child1]
    for (int idx = tid; idx < 16 * 288; idx += 256) {
      int r = idx / 288, k = (idx - r * 288) * 8;
      int node = tile * 16 + r;
      u32x4 val = {0u, 0u, 0u, 0u};
      if (node < levelN) {
        const __bf16* src = (k < HDIM)
            ? (h_bf + (size_t)(off + node) * HDIM + k)
            : (h_bf + (size_t)(choff + 2 * node) * HDIM + (k - HDIM));
        val = ldg((const u32x4*)src);
      }
      *(u32x4*)(ldsA + r * LDA + k) = val;
    }
  }
  __syncthreads();

  for (int jt = wave; jt < 48; jt += 8) {
    f32x8 acc[6];
#pragma unroll
    for (int s = 0; s < 6; ++s) { f32x8 z = {0.f,0.f,0.f,0.f,0.f,0.f,0.f,0.f}; acc[s] = z; }
    if (!leaf) {
      const __bf16* arow = ldsA + l * LDA;
      const __bf16* bp[6];
      size_t bstr[6];
      bp[0] = PWiou + (size_t)(jt)      * 512 + lane * 16; bstr[0] = (size_t)144 * 512;
      bp[1] = PWiou + (size_t)(48 + jt) * 512 + lane * 16; bstr[1] = (size_t)144 * 512;
      bp[2] = PWiou + (size_t)(96 + jt) * 512 + lane * 16; bstr[2] = (size_t)144 * 512;
      bp[3] = PWf0  + (size_t)jt        * 512 + lane * 16; bstr[3] = (size_t)48 * 512;
      bp[4] = PWf1  + (size_t)jt        * 512 + lane * 16; bstr[4] = (size_t)48 * 512;
      bp[5] = PWf2  + (size_t)jt        * 512 + lane * 16; bstr[5] = (size_t)48 * 512;
      for (int ks = 0; ks < 72; ++ks) {
        bf16x16 a = load_afrag(arow, ks * 32, hi);
#pragma unroll
        for (int s = 0; s < 6; ++s) {
          acc[s] = wmma_bf16(a, load_bfrag(bp[s]), acc[s]);
          bp[s] += bstr[s];
        }
      }
    }
    const int j = jt * 16 + l;
    const float bi = ldg(bt_iou + j), bo = ldg(bt_iou + HDIM + j), bg = ldg(bt_iou + 2 * HDIM + j);
    const float bfv = ldg(bt_f + j);
#pragma unroll
    for (int r = 0; r < 8; ++r) {
      int m = r + hi * 8;
      int node = tile * 16 + m;
      if (node < levelN) {
        int gid = off + node;
        int lid = ldg(ids + gid);
        const float* xr = Wt_iou + (size_t)lid * (3 * HDIM);
        float xf = ldg(Wt_f + (size_t)lid * HDIM + j);
        float ig = sigm(acc[0][r] + ldg(xr + j) + bi);
        float og = sigm(acc[1][r] + ldg(xr + HDIM + j) + bo);
        float gg = tanh_fast(acc[2][r] + ldg(xr + 2 * HDIM + j) + bg);
        float f0 = sigm(acc[3][r] + xf + bfv);  // self
        float f1 = sigm(acc[4][r] + xf + bfv);  // child0
        float f2 = sigm(acc[5][r] + xf + bfv);  // child1
        float ccS = 0.f, cc0 = 0.f, cc1 = 0.f;
        if (!leaf) {
          ccS = ldg(c_f32 + (size_t)(off + node) * HDIM + j);
          cc0 = ldg(c_f32 + (size_t)(choff + 2 * node) * HDIM + j);
          cc1 = ldg(c_f32 + (size_t)(choff + 2 * node + 1) * HDIM + j);
        }
        float cv = ig * gg + f0 * ccS + f1 * cc0 + f2 * cc1;
        float hv = og * tanh_fast(cv);
        stg(h3_bf + (size_t)gid * HDIM + j, (__bf16)hv);
      }
    }
  }
}

// ---------------- fused FFN + heads ----------------
__global__ void __launch_bounds__(256) ffn_heads(
    const __bf16* __restrict__ h3bf,
    const __bf16* __restrict__ PW1, const __bf16* __restrict__ PW2,
    const __bf16* __restrict__ PWH,
    const float* __restrict__ b1, const float* __restrict__ b2,
    const float* __restrict__ hl_b, const float* __restrict__ int_b,
    const float* __restrict__ act_b,
    const int* __restrict__ mask, float* __restrict__ out) {
  extern __shared__ __align__(16) char smem[];
  const int LDA = HDIM + 8;
  __bf16* ldsA = (__bf16*)smem;
  __bf16* ldsB = ldsA + 16 * LDA;
  float*  ldsS = (float*)(ldsB + 16 * LDA);
  const int tid = threadIdx.x, wave = tid >> 5, lane = tid & 31;
  const int l = lane & 15, hi = lane >> 4;
  const int base = blockIdx.x * 16;

  for (int idx = tid; idx < 16 * 96; idx += 256) {  // stage h3 tile
    int r = idx / 96, k = (idx - r * 96) * 8;
    int node = base + r;
    u32x4 val = {0u, 0u, 0u, 0u};
    if (node < NNODES) val = ldg((const u32x4*)(h3bf + (size_t)node * HDIM + k));
    *(u32x4*)(ldsA + r * LDA + k) = val;
  }
  __syncthreads();

  // GEMM1 + bias + relu -> ldsB (bf16)
  for (int jt = wave; jt < 48; jt += 8) {
    f32x8 acc = {0.f,0.f,0.f,0.f,0.f,0.f,0.f,0.f};
    const __bf16* bp = PW1 + (size_t)jt * 512 + lane * 16;
    const __bf16* arow = ldsA + l * LDA;
    for (int ks = 0; ks < 24; ++ks) {
      acc = wmma_bf16(load_afrag(arow, ks * 32, hi), load_bfrag(bp), acc);
      bp += (size_t)48 * 512;
    }
    int j = jt * 16 + l;
    float bb = ldg(b1 + j);
#pragma unroll
    for (int r = 0; r < 8; ++r) {
      float t = acc[r] + bb;
      ldsB[(r + hi * 8) * LDA + j] = (__bf16)(t > 0.f ? t : 0.f);
    }
  }
  __syncthreads();

  // GEMM2 + bias -> ldsA (bf16)
  for (int jt = wave; jt < 48; jt += 8) {
    f32x8 acc = {0.f,0.f,0.f,0.f,0.f,0.f,0.f,0.f};
    const __bf16* bp = PW2 + (size_t)jt * 512 + lane * 16;
    const __bf16* arow = ldsB + l * LDA;
    for (int ks = 0; ks < 24; ++ks) {
      acc = wmma_bf16(load_afrag(arow, ks * 32, hi), load_bfrag(bp), acc);
      bp += (size_t)48 * 512;
    }
    int j = jt * 16 + l;
    float bb = ldg(b2 + j);
#pragma unroll
    for (int r = 0; r < 8; ++r) ldsA[(r + hi * 8) * LDA + j] = (__bf16)(acc[r] + bb);
  }
  __syncthreads();

  // GEMM3: head scores 16x80 -> ldsS
  if (wave < 5) {
    f32x8 acc = {0.f,0.f,0.f,0.f,0.f,0.f,0.f,0.f};
    const __bf16* bp = PWH + (size_t)wave * 512 + lane * 16;
    const __bf16* arow = ldsA + l * LDA;
    for (int ks = 0; ks < 24; ++ks) {
      acc = wmma_bf16(load_afrag(arow, ks * 32, hi), load_bfrag(bp), acc);
      bp += (size_t)5 * 512;
    }
    int col = wave * 16 + l;
#pragma unroll
    for (int r = 0; r < 8; ++r) ldsS[(r + hi * 8) * 80 + col] = acc[r];
  }
  __syncthreads();

  // heads: one thread per node row
  if (tid < 16) {
    int node = base + tid;
    if (node < NNODES) {
      const float* srow = ldsS + tid * 80;
      const float invT = 1.f / 3.f;
      size_t ob = (size_t)node * 68;
      {
        float a0 = (srow[0] + ldg(hl_b + 0)) * invT, a1 = (srow[1] + ldg(hl_b + 1)) * invT;
        float mx = fmaxf(a0, a1);
        float e0 = __expf(a0 - mx), e1 = __expf(a1 - mx);
        float inv = 1.f / (e0 + e1);
        stg(out + ob + 0, e0 * inv); stg(out + ob + 1, e1 * inv);
      }
      {
        float a0 = (srow[2] + ldg(int_b + 0)) * invT, a1 = (srow[3] + ldg(int_b + 1)) * invT;
        float mx = fmaxf(a0, a1);
        float e0 = __expf(a0 - mx), e1 = __expf(a1 - mx);
        float inv = 1.f / (e0 + e1);
        stg(out + ob + 2, e0 * inv); stg(out + ob + 3, e1 * inv);
      }
      const int* mrow = mask + (size_t)node * 64;
      float mx = -1e30f;
      for (int q = 0; q < 64; ++q) {
        float lg = ldg(mrow + q) ? (srow[4 + q] + ldg(act_b + q)) * invT : (-1e30f * invT);
        mx = fmaxf(mx, lg);
      }
      float sum = 0.f;
      for (int q = 0; q < 64; ++q) {
        float lg = ldg(mrow + q) ? (srow[4 + q] + ldg(act_b + q)) * invT : (-1e30f * invT);
        float e = __expf(lg - mx);
        sum += e;
        stg(out + ob + 4 + q, e);
      }
      float inv = 1.f / sum;
      for (int q = 0; q < 64; ++q) stg(out + ob + 4 + q, ldg(out + ob + 4 + q) * inv);
    }
  }
}

// ---------------- host ----------------
extern "C" void kernel_launch(void* const* d_in, const int* in_sizes, int n_in,
                              void* d_out, int out_size, void* d_ws, size_t ws_size,
                              hipStream_t stream) {
  const int*   ids    = (const int*)d_in[0];
  const int*   mask   = (const int*)d_in[1];
  const float* Wb_iou = (const float*)d_in[2];
  const float* Ub_iou = (const float*)d_in[3];
  const float* bb_iou = (const float*)d_in[4];
  const float* Wb_f   = (const float*)d_in[5];
  const float* Ub_f   = (const float*)d_in[6];
  const float* bb_f   = (const float*)d_in[7];
  const float* Wt_iou = (const float*)d_in[8];
  const float* Ut_iou = (const float*)d_in[9];
  const float* bt_iou = (const float*)d_in[10];
  const float* Wt_f   = (const float*)d_in[11];
  const float* Ut_f   = (const float*)d_in[12];
  const float* bt_f   = (const float*)d_in[13];
  const float* fW1    = (const float*)d_in[14];
  const float* fb1    = (const float*)d_in[15];
  const float* fW2    = (const float*)d_in[16];
  const float* fb2    = (const float*)d_in[17];
  const float* hlW    = (const float*)d_in[18];
  const float* hlb    = (const float*)d_in[19];
  const float* inW    = (const float*)d_in[20];
  const float* inb    = (const float*)d_in[21];
  const float* acW    = (const float*)d_in[22];
  const float* acb    = (const float*)d_in[23];
  float* out = (float*)d_out;

  char* ws = (char*)d_ws;
  size_t o = 0;
  auto alloc = [&](size_t bytes) { size_t r = o; o += (bytes + 255) & ~(size_t)255; return r; };
  __bf16* h_bf   = (__bf16*)(ws + alloc((size_t)NNODES * HDIM * 2));
  float*  c_f32  = (float*) (ws + alloc((size_t)NNODES * HDIM * 4));
  __bf16* h3_bf  = (__bf16*)(ws + alloc((size_t)NNODES * HDIM * 2));
  __bf16* PW1iou = (__bf16*)(ws + alloc((size_t)1536 * 2304 * 2));
  __bf16* PW1f0  = (__bf16*)(ws + alloc((size_t)1536 * 768 * 2));
  __bf16* PW1f1  = (__bf16*)(ws + alloc((size_t)1536 * 768 * 2));
  __bf16* PW2iou = (__bf16*)(ws + alloc((size_t)2304 * 2304 * 2));
  __bf16* PW2f0  = (__bf16*)(ws + alloc((size_t)2304 * 768 * 2));
  __bf16* PW2f1  = (__bf16*)(ws + alloc((size_t)2304 * 768 * 2));
  __bf16* PW2f2  = (__bf16*)(ws + alloc((size_t)2304 * 768 * 2));
  __bf16* PWff1  = (__bf16*)(ws + alloc((size_t)768 * 768 * 2));
  __bf16* PWff2  = (__bf16*)(ws + alloc((size_t)768 * 768 * 2));
  __bf16* PWH    = (__bf16*)(ws + alloc((size_t)768 * 80 * 2));

  auto pk = [&](const float* src, __bf16* dst, int K, int O) {
    long total = (long)K * O;
    pack_w<<<(int)((total + 255) / 256), 256, 0, stream>>>(src, dst, K, O);
  };
  pk(Ub_iou, PW1iou, 1536, 2304);
  pk(Ub_f,                        PW1f0, 1536, 768);
  pk(Ub_f + (size_t)1536 * 768,   PW1f1, 1536, 768);
  pk(Ut_iou, PW2iou, 2304, 2304);
  pk(Ut_f,                          PW2f0, 2304, 768);
  pk(Ut_f + (size_t)2304 * 768,     PW2f1, 2304, 768);
  pk(Ut_f + (size_t)2 * 2304 * 768, PW2f2, 2304, 768);
  pk(fW1, PWff1, 768, 768);
  pk(fW2, PWff2, 768, 768);
  pack_heads<<<(768 * 80 + 255) / 256, 256, 0, stream>>>(hlW, inW, acW, PWH);

  const size_t sm1 = (size_t)16 * (1536 + 8) * 2;            // 49408
  const size_t sm2 = (size_t)16 * (2304 + 8) * 2;            // 73984
  const size_t smf = (size_t)16 * (HDIM + 8) * 2 * 2 + 16 * 80 * 4;  // 54784

  for (int d = DMAX; d >= 0; --d) {
    int tiles = ((1 << d) + 15) >> 4;
    cell_pass1<<<tiles, 256, sm1, stream>>>(ids, Wb_iou, Wb_f, bb_iou, bb_f,
                                            PW1iou, PW1f0, PW1f1, h_bf, c_f32, d);
  }

  int blocks2 = 0;
  for (int dd = 0; dd <= DMAX; ++dd) blocks2 += ((1 << dd) + 15) >> 4;
  cell_pass2<<<blocks2, 256, sm2, stream>>>(ids, Wt_iou, Wt_f, bt_iou, bt_f,
                                            PW2iou, PW2f0, PW2f1, PW2f2,
                                            h_bf, c_f32, h3_bf);

  ffn_heads<<<(NNODES + 15) / 16, 256, smf, stream>>>(h3_bf, PWff1, PWff2, PWH,
                                                      fb1, fb2, hlb, inb, acb, mask, out);
  (void)in_sizes; (void)n_in; (void)out_size; (void)ws_size;
}
